// AttentionFuserV2_65060164600006
// MI455X (gfx1250) — matrix operand order repaired and checked by reading, not compiled, not run on hardware
//
#include <hip/hip_runtime.h>
#include <hip/hip_bf16.h>

// Problem constants (match reference)
#define N_B 256
#define L_S 512
#define D_M 512

typedef __bf16 bf16;
typedef __attribute__((ext_vector_type(16))) __bf16 v16bf;
typedef __attribute__((ext_vector_type(8)))  float  v8f;
typedef __attribute__((ext_vector_type(4)))  unsigned int u32x4;
typedef __attribute__((ext_vector_type(8)))  int   i32x8;
typedef __attribute__((ext_vector_type(4)))  int   i32x4;

#if defined(__has_builtin)
#if __has_builtin(__builtin_amdgcn_tensor_load_to_lds) && \
    __has_builtin(__builtin_amdgcn_s_wait_tensorcnt)
#define HAVE_TDM 1
#endif
#endif

union ABFrag { uint4 u[2]; v16bf v; };

// ---------------------------------------------------------------------------
// WMMA fragment loaders (wave32, v_wmma_f32_16x16x32_bf16, contiguous k order k = 16h + s in A and B)
// ---------------------------------------------------------------------------

// A matrix 16x32 bf16, row-major source with row stride (elements).
// Lane 0-15 : row=lane,    K = kk+{0..15}
// Lane16-31 : row=lane-16, K = kk+{16..31}
__device__ __forceinline__ v16bf load_a_rm(const bf16* base, int rowStride,
                                           int kk, int lane) {
  const int r  = lane & 15;
  const int hi = (lane >> 4) & 1;
  const bf16* p = base + (size_t)r * rowStride + kk + hi * 16;
  ABFrag f;
  f.u[0] = *(const uint4*)(p);
  f.u[1] = *(const uint4*)(p + 8);
  return f.v;
}

// B matrix 32x16 bf16 from row-major M [N,K] (NT GEMM):
// lane n holds 16 contiguous K values of row n; lanes 16-31 hold K+16..31.
__device__ __forceinline__ v16bf load_b_nt(const bf16* base, int rowStride,
                                           int kk, int lane) {
  const int n  = lane & 15;
  const int hi = (lane >> 4) & 1;
  const bf16* p = base + (size_t)n * rowStride + kk + hi * 16;
  ABFrag f;
  f.u[0] = *(const uint4*)(p);
  f.u[1] = *(const uint4*)(p + 8);
  return f.v;
}

__device__ __forceinline__ v8f wmma_bf16(v16bf a, v16bf b, v8f c) {
  return __builtin_amdgcn_wmma_f32_16x16x32_bf16(
      false, a, false, b, (short)0, c, false, false);
}

// ---------------------------------------------------------------------------
// Utility kernels
// ---------------------------------------------------------------------------

__global__ void cvt_f32_to_bf16(const float* __restrict__ src,
                                bf16* __restrict__ dst, long n) {
  long stride = (long)gridDim.x * blockDim.x;
  for (long i = (long)blockIdx.x * blockDim.x + threadIdx.x; i < n; i += stride)
    dst[i] = (bf16)src[i];
}

// [b,l,d] -> [b,d,l]
__global__ void transpose_bf16(const bf16* __restrict__ src,
                               bf16* __restrict__ dst) {
  const long total = (long)N_B * L_S * D_M;
  long stride = (long)gridDim.x * blockDim.x;
  for (long i = (long)blockIdx.x * blockDim.x + threadIdx.x; i < total; i += stride) {
    int  d = (int)(i % D_M);
    long t = i / D_M;
    int  l = (int)(t % L_S);
    long b = t / L_S;
    dst[b * (long)D_M * L_S + (long)d * L_S + l] = src[i];
  }
}

// ---------------------------------------------------------------------------
// q = x @ W_in^T, bf16 out. One wave computes a 16x64 strip (A reused x4).
// ---------------------------------------------------------------------------
__global__ __launch_bounds__(128)
void gemm_nt_q(const bf16* __restrict__ X, const bf16* __restrict__ W,
               bf16* __restrict__ Q) {
  const int t    = blockIdx.x * 4 + (threadIdx.x >> 5);
  const int lane = threadIdx.x & 31;
  const int cb = t & 7;          // 64-col block
  const int r  = (t >> 3) & 31;  // 16-row tile
  const int b  = t >> 8;         // batch

  const bf16* Abase = X + (size_t)b * L_S * D_M + (size_t)(r * 16) * D_M;

  v8f acc[4] = {{}, {}, {}, {}};
  for (int kk = 0; kk < D_M; kk += 32) {
    if (kk + 64 < D_M)
      __builtin_prefetch(Abase + ((lane & 15) * D_M) + kk + 64, 0, 1);
    v16bf a = load_a_rm(Abase, D_M, kk, lane);
#pragma unroll
    for (int u = 0; u < 4; ++u) {
      v16bf bb = load_b_nt(W + (size_t)(cb * 64 + u * 16) * D_M, D_M, kk, lane);
      acc[u] = wmma_bf16(a, bb, acc[u]);
    }
  }

  const int n  = lane & 15;
  const int hi = (lane >> 4) & 1;
  bf16* out = Q + (size_t)b * L_S * D_M +
              (size_t)(r * 16 + hi * 8) * D_M + cb * 64 + n;
#pragma unroll
  for (int u = 0; u < 4; ++u)
#pragma unroll
    for (int i = 0; i < 8; ++i)
      out[(size_t)i * D_M + u * 16] = (bf16)acc[u][i];
}

// ---------------------------------------------------------------------------
// Fused scores + softmax + mix per (batch, 16-row query tile), 4 waves.
// q tile staged into LDS via TDM (tensor_load_to_lds); P aliases the q buffer.
// ---------------------------------------------------------------------------
__global__ __launch_bounds__(128)
void attn_kernel(const bf16* __restrict__ Q, const bf16* __restrict__ X,
                 const bf16* __restrict__ XT, bf16* __restrict__ MIX) {
  __shared__ __align__(16) bf16  sQP[16 * 512];  // q tile, later P (16 KB)
  __shared__ __align__(16) float sS[16 * 512];   // scores fp32    (32 KB)
  __shared__ float sred[16 * 8];
  __shared__ float srow[32];

  const int tid  = threadIdx.x;
  const int lane = tid & 31;
  const int wid  = tid >> 5;
  const int qt   = blockIdx.x & 31;
  const int b    = blockIdx.x >> 5;

  const bf16* qsrc = Q + (size_t)b * L_S * D_M + (size_t)qt * 16 * D_M;
  const bf16* Xb   = X + (size_t)b * L_S * D_M;

  // ---- stage A: q tile (16 KB, contiguous) -> LDS -------------------------
#if defined(HAVE_TDM)
  if (wid == 0) {
    const unsigned ldsa = (unsigned)(size_t)(void*)sQP;
    const unsigned long long ga = (unsigned long long)(size_t)qsrc;
    // D# group 0: count=1 | lds_addr | global_addr | type=2 (ISA 8.3)
    u32x4 g0;
    g0.x = 1u;
    g0.y = ldsa;
    g0.z = (unsigned)(ga & 0xFFFFFFFFu);
    g0.w = (unsigned)(((ga >> 32) & 0x1FFFFFFu) | (2u << 30));
    // D# group 1: data_size=2B; 1-D tile of 8192 elements (ISA 8.4)
    i32x8 g1;
    g1[0] = (int)(1u << 16);        // data_size = 1 -> 2 bytes
    g1[1] = (int)(0x2000u << 16);   // tensor_dim0 = 8192 (low 16)
    g1[2] = (int)(0x00010000u);     // tensor_dim0 hi=0 | tensor_dim1 = 1
    g1[3] = (int)(0x2000u << 16);   // tile_dim0 = 8192
    g1[4] = 0;                      // tile_dim1/2 unused
    g1[5] = 8192;                   // tensor_dim0_stride
    g1[6] = 0;
    g1[7] = 0;
    i32x4 z4 = {0, 0, 0, 0};
#if __clang_major__ >= 23
    i32x8 z8 = {0, 0, 0, 0, 0, 0, 0, 0};
    __builtin_amdgcn_tensor_load_to_lds(g0, g1, z4, z4, z8, 0);
#else
    __builtin_amdgcn_tensor_load_to_lds(g0, g1, z4, z4, 0);
#endif
    __builtin_amdgcn_s_wait_tensorcnt(0);
  }
#else
  for (int i = tid; i < (16 * 512) / 8; i += 128)
    ((uint4*)sQP)[i] = ((const uint4*)qsrc)[i];
#endif
  __syncthreads();

  // ---- stage B: scores, 2 key tiles per A fragment ------------------------
  for (int m0 = wid * 2; m0 < 32; m0 += 8) {
    v8f acc0 = {}, acc1 = {};
    const bf16* B0 = Xb + (size_t)(m0 * 16) * D_M;
    const bf16* B1 = Xb + (size_t)((m0 + 1) * 16) * D_M;
    for (int kk = 0; kk < D_M; kk += 32) {
      v16bf a = load_a_rm(sQP, 512, kk, lane);     // from LDS (ds_load_b128)
      acc0 = wmma_bf16(a, load_b_nt(B0, D_M, kk, lane), acc0);
      acc1 = wmma_bf16(a, load_b_nt(B1, D_M, kk, lane), acc1);
    }
    const int n  = lane & 15;
    const int hi = (lane >> 4) & 1;
#pragma unroll
    for (int i = 0; i < 8; ++i) {
      sS[(hi * 8 + i) * 512 + m0 * 16 + n]       = acc0[i];
      sS[(hi * 8 + i) * 512 + (m0 + 1) * 16 + n] = acc1[i];
    }
  }
  __syncthreads();

  // ---- softmax over 512 keys per row (16 rows x 8 chunks of 64) -----------
  const int row   = tid & 15;
  const int chunk = tid >> 4;
  {
    float mx = -3.4e38f;
    for (int j = chunk * 64; j < chunk * 64 + 64; ++j)
      mx = fmaxf(mx, sS[row * 512 + j]);
    sred[row * 8 + chunk] = mx;
  }
  __syncthreads();
  if (tid < 16) {
    float m2 = sred[tid * 8];
    for (int c = 1; c < 8; ++c) m2 = fmaxf(m2, sred[tid * 8 + c]);
    srow[tid] = m2;
  }
  __syncthreads();
  const float rm = srow[row];
  {
    float sm = 0.f;
    for (int j = chunk * 64; j < chunk * 64 + 64; ++j)
      sm += __expf(sS[row * 512 + j] - rm);
    sred[row * 8 + chunk] = sm;
  }
  __syncthreads();
  if (tid < 16) {
    float s2 = 0.f;
    for (int c = 0; c < 8; ++c) s2 += sred[tid * 8 + c];
    srow[16 + tid] = 1.0f / s2;
  }
  __syncthreads();
  const float inv = srow[16 + row];
  // q tile is dead now: overwrite sQP with P (bf16)
  for (int j = chunk * 64; j < chunk * 64 + 64; ++j)
    sQP[row * 512 + j] = (bf16)(__expf(sS[row * 512 + j] - rm) * inv);
  __syncthreads();

  // ---- stage D: mix = P @ x, 2 output tiles per A fragment ----------------
  const bf16* XTb = XT + (size_t)b * D_M * L_S;
  bf16* mout = MIX + (size_t)b * L_S * D_M + (size_t)qt * 16 * D_M;
  for (int j0 = wid * 2; j0 < 32; j0 += 8) {
    v8f acc0 = {}, acc1 = {};
    const bf16* B0 = XTb + (size_t)(j0 * 16) * L_S;
    const bf16* B1 = XTb + (size_t)((j0 + 1) * 16) * L_S;
    for (int kk = 0; kk < L_S; kk += 32) {
      v16bf a = load_a_rm(sQP, 512, kk, lane);     // P from LDS
      acc0 = wmma_bf16(a, load_b_nt(B0, L_S, kk, lane), acc0);
      acc1 = wmma_bf16(a, load_b_nt(B1, L_S, kk, lane), acc1);
    }
    const int n  = lane & 15;
    const int hi = (lane >> 4) & 1;
#pragma unroll
    for (int i = 0; i < 8; ++i) {
      mout[(size_t)(hi * 8 + i) * D_M + j0 * 16 + n]       = (bf16)acc0[i];
      mout[(size_t)(hi * 8 + i) * D_M + (j0 + 1) * 16 + n] = (bf16)acc1[i];
    }
  }
}

// ---------------------------------------------------------------------------
// out = [mix, q] @ W_out^T (K=1024 as two K=512 NT passes), 16x64 per wave.
// ---------------------------------------------------------------------------
__global__ __launch_bounds__(128)
void gemm_out(const bf16* __restrict__ MIXB, const bf16* __restrict__ QB,
              const bf16* __restrict__ W, bf16* __restrict__ dstb,
              float* __restrict__ dstf, int apply_tanh) {
  const int t    = blockIdx.x * 4 + (threadIdx.x >> 5);
  const int lane = threadIdx.x & 31;
  const int cb = t & 7;
  const int r  = (t >> 3) & 31;
  const int b  = t >> 8;

  const bf16* A1 = MIXB + (size_t)b * L_S * D_M + (size_t)(r * 16) * D_M;
  const bf16* A2 = QB   + (size_t)b * L_S * D_M + (size_t)(r * 16) * D_M;

  v8f acc[4] = {{}, {}, {}, {}};
  for (int kk = 0; kk < D_M; kk += 32) {           // mix half (c = 0..511)
    if (kk + 64 < D_M)
      __builtin_prefetch(A1 + ((lane & 15) * D_M) + kk + 64, 0, 1);
    v16bf a = load_a_rm(A1, D_M, kk, lane);
#pragma unroll
    for (int u = 0; u < 4; ++u) {
      v16bf bb = load_b_nt(W + (size_t)(cb * 64 + u * 16) * (2 * D_M),
                           2 * D_M, kk, lane);
      acc[u] = wmma_bf16(a, bb, acc[u]);
    }
  }
  for (int kk = 0; kk < D_M; kk += 32) {           // q half (c = 512..1023)
    v16bf a = load_a_rm(A2, D_M, kk, lane);
#pragma unroll
    for (int u = 0; u < 4; ++u) {
      v16bf bb = load_b_nt(W + (size_t)(cb * 64 + u * 16) * (2 * D_M),
                           2 * D_M, D_M + kk, lane);
      acc[u] = wmma_bf16(a, bb, acc[u]);
    }
  }

  const int n  = lane & 15;
  const int hi = (lane >> 4) & 1;
  const size_t base = (size_t)b * L_S * D_M +
                      (size_t)(r * 16 + hi * 8) * D_M + cb * 64 + n;
  if (apply_tanh) {
#pragma unroll
    for (int u = 0; u < 4; ++u)
#pragma unroll
      for (int i = 0; i < 8; ++i)
        dstb[base + (size_t)i * D_M + u * 16] = (bf16)tanhf(acc[u][i]);
  } else {
#pragma unroll
    for (int u = 0; u < 4; ++u)
#pragma unroll
      for (int i = 0; i < 8; ++i)
        dstf[base + (size_t)i * D_M + u * 16] = acc[u][i];
  }
}

// ---------------------------------------------------------------------------
// pooled[b,d] = mean over l of out[b,l,d]
// ---------------------------------------------------------------------------
__global__ void pool_kernel(const float* __restrict__ src,
                            float* __restrict__ out) {
  const int t = blockIdx.x * blockDim.x + threadIdx.x;
  if (t >= N_B * D_M) return;
  const int b = t / D_M;
  const int d = t - b * D_M;
  const float* p = src + (size_t)b * L_S * D_M + d;
  float s = 0.f;
  for (int l = 0; l < L_S; ++l) s += p[(size_t)l * D_M];
  out[t] = s * (1.0f / L_S);
}

// ---------------------------------------------------------------------------
// Host launch
// ---------------------------------------------------------------------------
extern "C" void kernel_launch(void* const* d_in, const int* in_sizes, int n_in,
                              void* d_out, int out_size, void* d_ws, size_t ws_size,
                              hipStream_t stream) {
  (void)in_sizes; (void)n_in; (void)out_size; (void)ws_size;
  const float* emb   = (const float*)d_in[0];
  const float* Win1  = (const float*)d_in[1];
  const float* Wout1 = (const float*)d_in[2];
  const float* Win2  = (const float*)d_in[3];
  const float* Wout2 = (const float*)d_in[4];

  const size_t NLD = (size_t)N_B * L_S * D_M;
  char* ws = (char*)d_ws;
  bf16*  xb    = (bf16*)ws;  ws += NLD * 2;   // current block input (bf16)
  bf16*  xT    = (bf16*)ws;  ws += NLD * 2;   // transposed input
  bf16*  qb    = (bf16*)ws;  ws += NLD * 2;   // q
  bf16*  mixb  = (bf16*)ws;  ws += NLD * 2;   // mix
  float* outf  = (float*)ws; ws += NLD * 4;   // final fp32 output
  bf16*  winb  = (bf16*)ws;  ws += (size_t)D_M * D_M * 2;
  bf16*  woutb = (bf16*)ws;

  const int gemmBlocks = (N_B * 32 * 8) / 4;    // 16384 blocks, 4 wave-strips
  const int attnBlocks = N_B * 32;              // 8192

  // ---- block 1 ----
  cvt_f32_to_bf16<<<4096, 256, 0, stream>>>(emb, xb, (long)NLD);
  cvt_f32_to_bf16<<<1024, 256, 0, stream>>>(Win1, winb, (long)D_M * D_M);
  cvt_f32_to_bf16<<<2048, 256, 0, stream>>>(Wout1, woutb, (long)D_M * 2 * D_M);
  transpose_bf16<<<4096, 256, 0, stream>>>(xb, xT);
  gemm_nt_q<<<gemmBlocks, 128, 0, stream>>>(xb, winb, qb);
  attn_kernel<<<attnBlocks, 128, 0, stream>>>(qb, xb, xT, mixb);
  gemm_out<<<gemmBlocks, 128, 0, stream>>>(mixb, qb, woutb, xb, outf, 1);

  // ---- block 2 ----
  cvt_f32_to_bf16<<<1024, 256, 0, stream>>>(Win2, winb, (long)D_M * D_M);
  cvt_f32_to_bf16<<<2048, 256, 0, stream>>>(Wout2, woutb, (long)D_M * 2 * D_M);
  transpose_bf16<<<4096, 256, 0, stream>>>(xb, xT);
  gemm_nt_q<<<gemmBlocks, 128, 0, stream>>>(xb, winb, qb);
  attn_kernel<<<attnBlocks, 128, 0, stream>>>(qb, xb, xT, mixb);
  gemm_out<<<gemmBlocks, 128, 0, stream>>>(mixb, qb, woutb, xb, outf, 0);

  // ---- mean pool over L ----
  pool_kernel<<<512, 256, 0, stream>>>(outf, (float*)d_out);
}
